// LongFo_60593398612322
// MI455X (gfx1250) — compile-verified, matrix operand order checked by reading, not run on hardware
//
#include <hip/hip_runtime.h>
#include <hip/hip_bf16.h>
#include <math.h>

// ---------------------------------------------------------------------------
// Longformer-style 2-layer encoder forward for MI455X (gfx1250, wave32).
// All matmuls via v_wmma_f32_16x16x32_bf16. Weights pre-transposed+converted
// to bf16 [N][K] (L2-resident: 192MB L2 >> 14MB/layer) so GEMM fragments are
// contiguous 16B loads and the GEMM inner loop is pure vmem + wmma.
// ---------------------------------------------------------------------------

#define HIDN 768
#define NHD  12
#define HDIM 64
#define NLYR 2
#define FFD  3072
#define SEQ  4096
#define WIN  256
#define NCHK (SEQ / WIN)
#define QKVN (3 * HIDN)          // 2304

typedef __bf16 bf16_t;
typedef __attribute__((ext_vector_type(16))) __bf16 v16bf;
typedef __attribute__((ext_vector_type(8)))  __bf16 v8bf;
typedef __attribute__((ext_vector_type(8)))  float  v8f;

__device__ __forceinline__ v16bf cat8(v8bf lo, v8bf hi) {
    return __builtin_shufflevector(lo, hi, 0, 1, 2, 3, 4, 5, 6, 7,
                                           8, 9, 10, 11, 12, 13, 14, 15);
}

__device__ __forceinline__ float red_max16(float x) {
#pragma unroll
    for (int m = 1; m < 16; m <<= 1) x = fmaxf(x, __shfl_xor(x, m, 32));
    return x;
}
__device__ __forceinline__ float red_sum16(float x) {
#pragma unroll
    for (int m = 1; m < 16; m <<= 1) x += __shfl_xor(x, m, 32);
    return x;
}

// ---------------------------------------------------------------------------
// Transpose + f32->bf16 convert: W[K][N] f32  ->  Wt[N][K] bf16.
// 32x32 LDS tile, coalesced reads and writes. Runs once per weight per call;
// output stays resident in the 192MB L2 for all GEMM blocks.
// ---------------------------------------------------------------------------
__global__ __launch_bounds__(256)
void wconv_kernel(const float* __restrict__ W, bf16_t* __restrict__ Wt,
                  int K, int N)
{
    __shared__ float tile[32][33];
    const int n0 = blockIdx.x * 32;
    const int k0 = blockIdx.y * 32;
    const int tx = threadIdx.x & 31;
    const int ty = threadIdx.x >> 5;   // 0..7
#pragma unroll
    for (int j = 0; j < 4; ++j)
        tile[ty + 8 * j][tx] = W[(k0 + ty + 8 * j) * N + n0 + tx];
    __syncthreads();
#pragma unroll
    for (int j = 0; j < 4; ++j)
        Wt[(n0 + ty + 8 * j) * K + k0 + tx] = (bf16_t)tile[tx][ty + 8 * j];
}

__global__ __launch_bounds__(256)
void pack_bias_kernel(const float* __restrict__ a, const float* __restrict__ b,
                      const float* __restrict__ c, float* __restrict__ out)
{
    int i = blockIdx.x * 256 + threadIdx.x;   // grid 3 x 256 = 768
    out[i]            = a[i];
    out[HIDN + i]     = b[i];
    out[2 * HIDN + i] = c[i];
}

// ---------------------------------------------------------------------------
// GEMM: out = act(A_bf16[MxK] @ Bt_bf16[NxK]^T + bias) (+ residual)
// Block 128x256 (8 waves as 2x4), wave tile 64x64 (16 WMMA accumulators),
// K-step 32. No LDS, no barriers: pure b128 loads (L2) + wmma, fully
// software-pipelineable. ~32 FLOP/byte from L2 per wave.
// ---------------------------------------------------------------------------
__global__ __launch_bounds__(256)
void gemm_bf16_kernel(const bf16_t* __restrict__ A, const bf16_t* __restrict__ Bt,
                      const float* __restrict__ bias, const float* __restrict__ res,
                      float* __restrict__ outF, bf16_t* __restrict__ outB,
                      int M, int N, int K, int do_gelu)
{
    const int tid  = threadIdx.x;
    const int lane = tid & 31;
    const int wv   = tid >> 5;
    const int wm   = wv >> 2;              // 0..1
    const int wn   = wv & 3;               // 0..3
    const int bm   = blockIdx.y * 128;
    const int bn   = blockIdx.x * 256;
    const int l16  = lane & 15;
    const int kb8  = (lane >> 4) << 3;     // 0 or 8

    v8f zf = {0.f, 0.f, 0.f, 0.f, 0.f, 0.f, 0.f, 0.f};
    v8f acc[4][4];
#pragma unroll
    for (int i = 0; i < 4; ++i)
#pragma unroll
        for (int j = 0; j < 4; ++j) acc[i][j] = zf;

    // 32-bit base offsets (all < 2^31)
    const int arow0 = (bm + wm * 64 + l16) * K + kb8;
    const int brow0 = (bn + wn * 64 + l16) * K + kb8;

    for (int k0 = 0; k0 < K; k0 += 32) {
        v16bf af[4], bfm[4];
#pragma unroll
        for (int mt = 0; mt < 4; ++mt) {
            const bf16_t* ap = A + arow0 + mt * 16 * K + k0;
            af[mt] = cat8(*(const v8bf*)ap, *(const v8bf*)(ap + 16));
        }
#pragma unroll
        for (int nt = 0; nt < 4; ++nt) {
            const bf16_t* bp = Bt + brow0 + nt * 16 * K + k0;
            bfm[nt] = cat8(*(const v8bf*)bp, *(const v8bf*)(bp + 16));
        }
#pragma unroll
        for (int mt = 0; mt < 4; ++mt)
#pragma unroll
            for (int nt = 0; nt < 4; ++nt)
                acc[mt][nt] = __builtin_amdgcn_wmma_f32_16x16x32_bf16(
                    false, af[mt], false, bfm[nt], (short)0, acc[mt][nt], false, false);
    }

    const float inv_sqrt2 = 0.70710678118654752440f;
#pragma unroll
    for (int mt = 0; mt < 4; ++mt) {
#pragma unroll
        for (int nt = 0; nt < 4; ++nt) {
#pragma unroll
            for (int r = 0; r < 8; ++r) {
                int row = bm + wm * 64 + mt * 16 + r + ((lane < 16) ? 0 : 8);
                int col = bn + wn * 64 + nt * 16 + l16;
                float v = acc[mt][nt][r] + bias[col];
                if (do_gelu) v = 0.5f * v * (1.0f + erff(v * inv_sqrt2));
                if (res)     v += res[row * N + col];
                int o = row * N + col;
                if (outF) outF[o] = v;
                if (outB) outB[o] = (bf16_t)v;
            }
        }
    }
}

// ---------------------------------------------------------------------------
// Sliding-window flash attention: one block per (chunk, head); 8 waves,
// each wave owns 32 query rows (2 M-tiles), streams 24 key blocks of 32.
// QKV read from packed [S][2304] bf16 buffer (q | k | v).
// ---------------------------------------------------------------------------
__global__ __launch_bounds__(256)
void attn_kernel(const bf16_t* __restrict__ qkv, const int* __restrict__ mask,
                 bf16_t* __restrict__ ob)
{
    __shared__ bf16_t vt[64][40];        // V^T block: [dim][key]
    __shared__ bf16_t pl[8][16][40];     // per-wave P-tile staging

    const int c    = blockIdx.x;         // chunk
    const int h    = blockIdx.y;         // head
    const int tid  = threadIdx.x;
    const int lane = tid & 31;
    const int wv   = tid >> 5;
    const int l16  = lane & 15;
    const int kb8  = (lane >> 4) << 3;
    const int mr8  = (lane < 16) ? 0 : 8;
    const int qbase = wv * 32;
    const int kwin0 = c * WIN - WIN;     // absolute position of window start
    const float scale = 0.125f;          // 1/sqrt(64)

    // preload Q fragments (reused for all 24 key blocks)
    v16bf aq[2][2];
#pragma unroll
    for (int mt = 0; mt < 2; ++mt) {
        int tok = c * WIN + qbase + mt * 16 + l16;
#pragma unroll
        for (int ks = 0; ks < 2; ++ks) {
            const bf16_t* p = qkv + tok * QKVN + h * HDIM + ks * 32 + kb8;
            aq[mt][ks] = cat8(*(const v8bf*)p, *(const v8bf*)(p + 16));
        }
    }

    v8f zf = {0.f, 0.f, 0.f, 0.f, 0.f, 0.f, 0.f, 0.f};
    v8f o[2][4];
    float mM[2][8], lM[2][8];
#pragma unroll
    for (int mt = 0; mt < 2; ++mt) {
#pragma unroll
        for (int n = 0; n < 4; ++n) o[mt][n] = zf;
#pragma unroll
        for (int r = 0; r < 8; ++r) { mM[mt][r] = -1e30f; lM[mt][r] = 0.f; }
    }

    for (int blk = 0; blk < 24; ++blk) {
        const int j0 = blk * 32;
        // stage V^T: one 16B load per thread (256 threads = 32 keys x 8 dims)
        {
            int kk = tid >> 3;           // 0..31 key within block
            int d0 = (tid & 7) * 8;      // 0..56 dim group
            int ka = kwin0 + j0 + kk;
            v8bf vvv = {};
            if (ka >= 0 && ka < SEQ)
                vvv = *(const v8bf*)(qkv + ka * QKVN + 2 * HIDN + h * HDIM + d0);
#pragma unroll
            for (int j2 = 0; j2 < 8; ++j2) vt[d0 + j2][kk] = vvv[j2];
        }
        __syncthreads();

        // wave-uniform band culling (barriers stay uniform across waves)
        bool active = (j0 + 31 >= qbase) && (j0 <= qbase + 31 + 2 * WIN);
        if (active) {
            // K fragments (B of Q*K^T), guarded per-lane
            v16bf bk[2][2];
            float kmsk[2];
#pragma unroll
            for (int nt = 0; nt < 2; ++nt) {
                int j  = j0 + nt * 16 + l16;
                int ka = kwin0 + j;
                bool inr = (ka >= 0) && (ka < SEQ);
                kmsk[nt] = (inr && mask[inr ? ka : 0] != 0) ? 1.0f : 0.0f;
#pragma unroll
                for (int ks = 0; ks < 2; ++ks) {
                    if (inr) {
                        const bf16_t* p = qkv + ka * QKVN + HIDN + h * HDIM + ks * 32 + kb8;
                        bk[nt][ks] = cat8(*(const v8bf*)p, *(const v8bf*)(p + 16));
                    } else {
                        v8bf z = {};
                        bk[nt][ks] = cat8(z, z);
                    }
                }
            }
            // V fragments (B of P*V) from transposed LDS
            v16bf bv[4];
#pragma unroll
            for (int n = 0; n < 4; ++n) {
                const bf16_t* p = &vt[n * 16 + l16][kb8];
                bv[n] = cat8(*(const v8bf*)p, *(const v8bf*)(p + 16));
            }

#pragma unroll
            for (int mt = 0; mt < 2; ++mt) {
                v8f s0 = zf, s1 = zf;
                s0 = __builtin_amdgcn_wmma_f32_16x16x32_bf16(false, aq[mt][0], false, bk[0][0], (short)0, s0, false, false);
                s0 = __builtin_amdgcn_wmma_f32_16x16x32_bf16(false, aq[mt][1], false, bk[0][1], (short)0, s0, false, false);
                s1 = __builtin_amdgcn_wmma_f32_16x16x32_bf16(false, aq[mt][0], false, bk[1][0], (short)0, s1, false, false);
                s1 = __builtin_amdgcn_wmma_f32_16x16x32_bf16(false, aq[mt][1], false, bk[1][1], (short)0, s1, false, false);

                float sc0[8], sc1[8];
#pragma unroll
                for (int r = 0; r < 8; ++r) {
                    int q  = qbase + mt * 16 + r + mr8;
                    int jA = j0 + l16;
                    int jB = j0 + 16 + l16;
                    bool vA = (jA >= q) && (jA <= q + 2 * WIN) && (kmsk[0] > 0.5f);
                    bool vB = (jB >= q) && (jB <= q + 2 * WIN) && (kmsk[1] > 0.5f);
                    sc0[r] = vA ? s0[r] * scale : -1e9f;
                    sc1[r] = vB ? s1[r] * scale : -1e9f;
                }
                // online softmax
#pragma unroll
                for (int r = 0; r < 8; ++r) {
                    float cur = red_max16(fmaxf(sc0[r], sc1[r]));
                    float mn  = fmaxf(mM[mt][r], cur);
                    float fac = __expf(mM[mt][r] - mn);
                    float p0  = __expf(sc0[r] - mn);
                    float p1  = __expf(sc1[r] - mn);
                    lM[mt][r] = lM[mt][r] * fac + red_sum16(p0 + p1);
                    mM[mt][r] = mn;
#pragma unroll
                    for (int n = 0; n < 4; ++n) o[mt][n][r] *= fac;
                    sc0[r] = p0; sc1[r] = p1;
                }
                // C-layout -> A-fragment layout via per-wave LDS round trip
#pragma unroll
                for (int r = 0; r < 8; ++r) {
                    pl[wv][r + mr8][l16]      = (bf16_t)sc0[r];
                    pl[wv][r + mr8][16 + l16] = (bf16_t)sc1[r];
                }
                asm volatile("s_wait_dscnt 0" ::: "memory");
                v16bf ap;
                {
                    const bf16_t* p = &pl[wv][l16][kb8];
                    ap = cat8(*(const v8bf*)p, *(const v8bf*)(p + 16));
                }
#pragma unroll
                for (int n = 0; n < 4; ++n)
                    o[mt][n] = __builtin_amdgcn_wmma_f32_16x16x32_bf16(
                        false, ap, false, bv[n], (short)0, o[mt][n], false, false);
            }
        }
        __syncthreads();
    }

    // normalize and write bf16 output in [S][HID] layout
#pragma unroll
    for (int mt = 0; mt < 2; ++mt)
#pragma unroll
        for (int n = 0; n < 4; ++n)
#pragma unroll
            for (int r = 0; r < 8; ++r) {
                int tok = c * WIN + qbase + mt * 16 + r + mr8;
                int col = h * HDIM + n * 16 + l16;
                float val = o[mt][n][r] / fmaxf(lM[mt][r], 1e-20f);
                ob[tok * HIDN + col] = (bf16_t)val;
            }
}

// ---------------------------------------------------------------------------
// LayerNorm over rows of 768; writes f32 (+ optional bf16 copy)
// ---------------------------------------------------------------------------
__global__ __launch_bounds__(256)
void layernorm_kernel(const float* __restrict__ in, const float* __restrict__ g,
                      const float* __restrict__ b, float* __restrict__ outF,
                      bf16_t* __restrict__ outB)
{
    __shared__ float red[16];
    const int row = blockIdx.x;
    const int tid = threadIdx.x;
    const float* x = in + row * HIDN;
    float v0 = x[tid], v1 = x[tid + 256], v2 = x[tid + 512];
    float s  = v0 + v1 + v2;
    float s2 = v0 * v0 + v1 * v1 + v2 * v2;
#pragma unroll
    for (int m = 1; m < 32; m <<= 1) { s += __shfl_xor(s, m, 32); s2 += __shfl_xor(s2, m, 32); }
    int wv = tid >> 5, lane = tid & 31;
    if (lane == 0) { red[wv] = s; red[8 + wv] = s2; }
    __syncthreads();
    if (tid == 0) {
        float ts = 0.f, ts2 = 0.f;
        for (int i = 0; i < 8; ++i) { ts += red[i]; ts2 += red[8 + i]; }
        red[0] = ts; red[8] = ts2;
    }
    __syncthreads();
    float mean = red[0] * (1.0f / HIDN);
    float var  = red[8] * (1.0f / HIDN) - mean * mean;
    float rstd = rsqrtf(var + 1e-5f);
#pragma unroll
    for (int i = 0; i < 3; ++i) {
        int cidx = tid + i * 256;
        float val = (x[cidx] - mean) * rstd * g[cidx] + b[cidx];
        outF[row * HIDN + cidx] = val;
        if (outB) outB[row * HIDN + cidx] = (bf16_t)val;
    }
}

// ---------------------------------------------------------------------------
// Position ids (cumsum of mask) and embedding gather/sum
// ---------------------------------------------------------------------------
__global__ void posids_kernel(const int* __restrict__ mask, int* __restrict__ pos)
{
    if (threadIdx.x == 0 && blockIdx.x == 0) {
        int csum = 0;
        for (int i = 0; i < SEQ; ++i) {
            int m = mask[i];
            csum += m;
            pos[i] = csum * m + 1;
        }
    }
}

__global__ __launch_bounds__(256)
void embed_kernel(const int* __restrict__ ip, const int* __restrict__ pos,
                  const float* __restrict__ we, const float* __restrict__ pe,
                  const float* __restrict__ te, float* __restrict__ out)
{
    const int row = blockIdx.x;
    const int tid = threadIdx.x;
    const int wid = ip[row];
    const int pid = pos[row];
#pragma unroll
    for (int i = 0; i < 3; ++i) {
        int cidx = tid + i * 256;
        out[row * HIDN + cidx] = we[wid * HIDN + cidx] + pe[pid * HIDN + cidx] + te[cidx];
    }
}

// ---------------------------------------------------------------------------
// Launcher
// ---------------------------------------------------------------------------
extern "C" void kernel_launch(void* const* d_in, const int* in_sizes, int n_in,
                              void* d_out, int out_size, void* d_ws, size_t ws_size,
                              hipStream_t stream)
{
    const int*   ip       = (const int*)d_in[0];
    const int*   mask     = (const int*)d_in[1];
    const float* word_emb = (const float*)d_in[2];
    const float* pos_emb  = (const float*)d_in[3];
    const float* type_emb = (const float*)d_in[4];
    const float* ln_emb_g = (const float*)d_in[5];
    const float* ln_emb_b = (const float*)d_in[6];
    const float* Wq = (const float*)d_in[7];
    const float* bq = (const float*)d_in[8];
    const float* Wk = (const float*)d_in[9];
    const float* bk = (const float*)d_in[10];
    const float* Wv = (const float*)d_in[11];
    const float* bv = (const float*)d_in[12];
    const float* Wo = (const float*)d_in[13];
    const float* bo = (const float*)d_in[14];
    const float* ln1_g = (const float*)d_in[15];
    const float* ln1_b = (const float*)d_in[16];
    const float* W1 = (const float*)d_in[17];
    const float* b1 = (const float*)d_in[18];
    const float* W2 = (const float*)d_in[19];
    const float* b2 = (const float*)d_in[20];
    const float* ln2_g = (const float*)d_in[21];
    const float* ln2_b = (const float*)d_in[22];

    size_t off = 0;
    char* base = (char*)d_ws;
    auto carve = [&](size_t bytes) -> void* {
        void* p = base + off;
        off += (bytes + 255) & ~(size_t)255;
        return p;
    };
    int*    posb  = (int*)   carve((size_t)SEQ * sizeof(int));
    float*  x     = (float*) carve((size_t)SEQ * HIDN * sizeof(float));
    float*  t     = (float*) carve((size_t)SEQ * HIDN * sizeof(float));
    bf16_t* xb    = (bf16_t*)carve((size_t)SEQ * HIDN * sizeof(bf16_t));
    bf16_t* qkvbuf= (bf16_t*)carve((size_t)SEQ * QKVN * sizeof(bf16_t));
    bf16_t* abuf  = (bf16_t*)carve((size_t)SEQ * HIDN * sizeof(bf16_t));
    bf16_t* hbuf  = (bf16_t*)carve((size_t)SEQ * FFD  * sizeof(bf16_t));
    bf16_t* wqkvT = (bf16_t*)carve((size_t)QKVN * HIDN * sizeof(bf16_t));
    bf16_t* woT   = (bf16_t*)carve((size_t)HIDN * HIDN * sizeof(bf16_t));
    bf16_t* w1T   = (bf16_t*)carve((size_t)FFD  * HIDN * sizeof(bf16_t));
    bf16_t* w2T   = (bf16_t*)carve((size_t)HIDN * FFD  * sizeof(bf16_t));
    float*  qkvb  = (float*) carve((size_t)QKVN * sizeof(float));
    (void)ws_size; (void)in_sizes; (void)n_in; (void)out_size;

    const dim3 blk(256);
    const dim3 gQKV(QKVN / 256, SEQ / 128);   // (9, 32)
    const dim3 gHID(HIDN / 256, SEQ / 128);   // (3, 32)
    const dim3 gFF (FFD  / 256, SEQ / 128);   // (12, 32)
    const dim3 gAttn(NCHK, NHD);              // (16, 12)
    const dim3 gT88(HIDN / 32, HIDN / 32);    // 768x768 transpose
    const dim3 gT8F(FFD  / 32, HIDN / 32);    // W1: K=768,N=3072
    const dim3 gTF8(HIDN / 32, FFD  / 32);    // W2: K=3072,N=768

    // embeddings + LN
    posids_kernel<<<1, 32, 0, stream>>>(mask, posb);
    embed_kernel<<<SEQ, blk, 0, stream>>>(ip, posb, word_emb, pos_emb, type_emb, t);
    layernorm_kernel<<<SEQ, blk, 0, stream>>>(t, ln_emb_g, ln_emb_b, x, xb);

    for (int l = 0; l < NLYR; ++l) {
        const int wo88 = l * HIDN * HIDN;
        const int wo8F = l * HIDN * FFD;

        // convert weights: bf16, transposed [N][K]; QKV packed into one matrix
        wconv_kernel<<<gT88, blk, 0, stream>>>(Wq + wo88, wqkvT,                    HIDN, HIDN);
        wconv_kernel<<<gT88, blk, 0, stream>>>(Wk + wo88, wqkvT + HIDN * HIDN,      HIDN, HIDN);
        wconv_kernel<<<gT88, blk, 0, stream>>>(Wv + wo88, wqkvT + 2 * HIDN * HIDN,  HIDN, HIDN);
        wconv_kernel<<<gT88, blk, 0, stream>>>(Wo + wo88, woT,                      HIDN, HIDN);
        wconv_kernel<<<gT8F, blk, 0, stream>>>(W1 + wo8F, w1T,                      HIDN, FFD);
        wconv_kernel<<<gTF8, blk, 0, stream>>>(W2 + wo8F, w2T,                      FFD,  HIDN);
        pack_bias_kernel<<<3, blk, 0, stream>>>(bq + l * HIDN, bk + l * HIDN, bv + l * HIDN, qkvb);

        // fused QKV projection -> packed bf16 [S][2304]
        gemm_bf16_kernel<<<gQKV, blk, 0, stream>>>(xb, wqkvT, qkvb,
            nullptr, nullptr, qkvbuf, SEQ, QKVN, HIDN, 0);

        // sliding-window attention
        attn_kernel<<<gAttn, blk, 0, stream>>>(qkvbuf, mask, abuf);

        // output projection + residual, then LN1
        gemm_bf16_kernel<<<gHID, blk, 0, stream>>>(abuf, woT, bo + l * HIDN,
            x, t, nullptr, SEQ, HIDN, HIDN, 0);
        layernorm_kernel<<<SEQ, blk, 0, stream>>>(t, ln1_g + l * HIDN, ln1_b + l * HIDN, x, xb);

        // FFN: gelu(x@W1+b1) -> bf16, then @W2+b2 + residual, LN2
        gemm_bf16_kernel<<<gFF, blk, 0, stream>>>(xb, w1T, b1 + l * FFD,
            nullptr, nullptr, hbuf, SEQ, FFD, HIDN, 1);
        gemm_bf16_kernel<<<gHID, blk, 0, stream>>>(hbuf, w2T, b2 + l * HIDN,
            x, t, nullptr, SEQ, HIDN, FFD, 0);

        float* xdst = (l == NLYR - 1) ? (float*)d_out : x;
        layernorm_kernel<<<SEQ, blk, 0, stream>>>(t, ln2_g + l * HIDN, ln2_b + l * HIDN, xdst, xb);
    }
}
